// NormEMAVectorQuantizer_9070970929190
// MI455X (gfx1250) — matrix operand order repaired and checked by reading, not compiled, not run on hardware
//
#include <hip/hip_runtime.h>

typedef __attribute__((ext_vector_type(16))) __bf16 v16bf;
typedef __attribute__((ext_vector_type(8)))  float  v8f;

constexpr int   D    = 128;     // channel / feature dim
constexpr int   K    = 8192;    // codebook size
constexpr int   NQ   = 16384;   // B*H*W query vectors
constexpr int   HW   = 1024;    // H*W
constexpr float BETA = 0.25f;

constexpr int   NSEG = 2;            // K split across blocks
constexpr int   KTILES_SEG = (K / 16) / NSEG;   // 256 k-tiles per segment

__device__ __forceinline__ unsigned short f32_to_bf16(float f) {
    unsigned int u = __float_as_uint(f);
    unsigned int r = u + 0x7FFFu + ((u >> 16) & 1u);   // round-to-nearest-even
    return (unsigned short)(r >> 16);
}

// CDNA5 async global->LDS copy (ASYNCcnt-tracked), GV addressing mode.
__device__ __forceinline__ void async_load_b128(unsigned lds_byte_off,
                                                const void* gaddr) {
    asm volatile("global_load_async_to_lds_b128 %0, %1, off"
                 :: "v"(lds_byte_off), "v"(gaddr) : "memory");
}
#define WAIT_ASYNC(n) asm volatile("s_wait_asynccnt " #n ::: "memory")

// ---------------------------------------------------------------------------
// Kernel 1: codebook fp32 -> bf16 + per-row squared norm
// ---------------------------------------------------------------------------
__global__ __launch_bounds__(128)
void conv_weight(const float* __restrict__ w, unsigned short* __restrict__ wbf,
                 float* __restrict__ csq) {
    int k = blockIdx.x, c = threadIdx.x;
    float v = w[(size_t)k * D + c];
    wbf[(size_t)k * D + c] = f32_to_bf16(v);
    __shared__ float s[128];
    s[c] = v * v;
    __syncthreads();
    for (int o = 64; o > 0; o >>= 1) {
        if (c < o) s[c] += s[c + o];
        __syncthreads();
    }
    if (c == 0) csq[k] = s[0];
}

// ---------------------------------------------------------------------------
// Kernel 2: gather z[b,:,h,w], L2-normalize, emit bf16 (N x 128). Zero loss.
// ---------------------------------------------------------------------------
__global__ __launch_bounds__(128)
void norm_z(const float* __restrict__ z, unsigned short* __restrict__ zbf,
            float* __restrict__ loss) {
    int n = blockIdx.x, c = threadIdx.x;
    if (n == 0 && c == 0) *loss = 0.0f;
    int b = n / HW, hw = n % HW;
    size_t off = (size_t)b * D * HW + (size_t)c * HW + hw;  // (B,C,H,W) layout
    float v = z[off];
    __shared__ float s[128];
    s[c] = v * v;
    __syncthreads();
    for (int o = 64; o > 0; o >>= 1) {
        if (c < o) s[c] += s[c + o];
        __syncthreads();
    }
    float inv = 1.0f / fmaxf(sqrtf(s[0]), 1e-12f);
    zbf[(size_t)n * D + c] = f32_to_bf16(v * inv);
}

// ---------------------------------------------------------------------------
// Kernel 3: WMMA distance GEMM + argmin over one K segment.
// Block = 256 threads = 8 waves covering 128 query rows; the 16-codeword B
// tile (4 KB) is async-copied global->LDS once per k-tile (double buffered)
// and shared by all 8 waves.  dist = |c|^2 - 2*dot (row |x|^2 irrelevant).
// ---------------------------------------------------------------------------
__global__ __launch_bounds__(256)
void argmin_wmma(const unsigned short* __restrict__ zbf,
                 const unsigned short* __restrict__ wbf,
                 const float* __restrict__ csq,
                 float* __restrict__ bestv_out,
                 int*   __restrict__ besti_out) {
    __shared__ __align__(16) unsigned short tileB[2][16 * D];   // 2 x 4 KB

    int tid  = threadIdx.x;
    int lane = tid & 31;
    int wave = tid >> 5;
    int g    = blockIdx.x >> 1;          // query row-group (128 rows)
    int seg  = blockIdx.x & 1;           // codebook segment
    int kbase = seg * (K / NSEG);        // first codeword of segment

    // ---- A fragments: 16-bit A layout, resident for the whole sweep -------
    int n0   = g * 128 + wave * 16;
    int m    = lane & 15;
    int half = lane >> 4;
    union Frag { v16bf v; uint4 q[2]; };
    Frag a[4];
    const uint4* zrow = (const uint4*)(zbf + (size_t)(n0 + m) * D);
#pragma unroll
    for (int f = 0; f < 4; ++f) {
        int e0 = 32 * f + 8 * half;            // ushort index within row
        a[f].q[0] = zrow[e0 >> 3];
        a[f].q[1] = zrow[(e0 + 16) >> 3];
    }

    // ---- async-copy plumbing: one b128 per thread per k-tile --------------
    int r      = tid >> 4;                 // codeword row within tile (0..15)
    int chunk  = tid & 15;                 // 16-byte chunk within row
    const unsigned short* gbase =
        wbf + (size_t)(kbase + r) * D + chunk * 8;
    unsigned lds0 = (unsigned)(uintptr_t)(&tileB[0][0]) + (unsigned)tid * 16;

    float best[8];
    int   bidx[8];
#pragma unroll
    for (int j = 0; j < 8; ++j) { best[j] = 3.4e38f; bidx[j] = 0; }

    int ncol = lane & 15;                  // this lane's C/D column

    async_load_b128(lds0, gbase);          // prologue: tile 0 -> buf 0

    for (int kt = 0; kt < KTILES_SEG; ++kt) {
        int cur = kt & 1;
        if (kt + 1 < KTILES_SEG) {
            async_load_b128(lds0 + (unsigned)((kt + 1) & 1) * (16 * D * 2),
                            gbase + (size_t)(kt + 1) * 16 * D);
            WAIT_ASYNC(1);                 // tile kt landed (in-order)
        } else {
            WAIT_ASYNC(0);
        }
        __syncthreads();                   // all waves' chunks visible

        // B fragments from LDS: column ncol = codeword row, K-elements
        // 8*half..+7 and 16+8*half..+7 per 32-wide K window.
        const uint4* bp = (const uint4*)&tileB[cur][ncol * D];
        Frag bf[4];
#pragma unroll
        for (int f = 0; f < 4; ++f) {
            bf[f].q[0] = bp[4 * f + half];
            bf[f].q[1] = bp[4 * f + 2 + half];
        }
        v8f acc = {};
#pragma unroll
        for (int f = 0; f < 4; ++f)
            acc = __builtin_amdgcn_wmma_f32_16x16x32_bf16(
                false, a[f].v, false, bf[f].v, (short)0, acc, false, false);

        int   krow = kbase + kt * 16 + ncol;
        float cs   = csq[krow];
#pragma unroll
        for (int j = 0; j < 8; ++j) {
            float d = cs - 2.0f * acc[j];
            if (d < best[j]) { best[j] = d; bidx[j] = krow; }
        }
        __syncthreads();                   // done reading buf cur
    }

    // cross-lane min-reduce within each 16-lane half; row = j + 8*half
#pragma unroll
    for (int j = 0; j < 8; ++j) {
        float bv = best[j];
        int   bi = bidx[j];
        for (int off = 8; off > 0; off >>= 1) {
            float ov = __shfl_xor(bv, off, 16);
            int   oi = __shfl_xor(bi, off, 16);
            if (ov < bv || (ov == bv && oi < bi)) { bv = ov; bi = oi; }
        }
        if ((lane & 15) == 0) {
            int row = n0 + j + 8 * half;
            bestv_out[(size_t)seg * NQ + row] = bv;
            besti_out[(size_t)seg * NQ + row] = bi;
        }
    }
}

// ---------------------------------------------------------------------------
// Kernel 4: combine segments, z_q scatter (B,C,H,W), loss, indices-as-float
// ---------------------------------------------------------------------------
__global__ __launch_bounds__(128)
void output_kernel(const float* __restrict__ z, const float* __restrict__ w,
                   const float* __restrict__ bestv, const int* __restrict__ besti,
                   float* __restrict__ zq_out, float* __restrict__ loss,
                   float* __restrict__ idxf) {
    int n = blockIdx.x, c = threadIdx.x;
    // 2-way argmin combine; <= keeps the lower-index segment on ties.
    float v0 = bestv[n], v1 = bestv[NQ + n];
    int   k  = (v0 <= v1) ? besti[n] : besti[NQ + n];

    int b = n / HW, hw = n % HW;
    size_t off = (size_t)b * D * HW + (size_t)c * HW + hw;
    float v = z[off];
    __shared__ float s[128];
    s[c] = v * v;
    __syncthreads();
    for (int o = 64; o > 0; o >>= 1) {
        if (c < o) s[c] += s[c + o];
        __syncthreads();
    }
    float zt = v * (1.0f / fmaxf(sqrtf(s[0]), 1e-12f));
    float wv = w[(size_t)k * D + c];
    zq_out[off] = wv;                 // forward value of straight-through z_q
    float dfr = wv - zt;
    __syncthreads();
    s[c] = dfr * dfr;
    __syncthreads();
    for (int o = 64; o > 0; o >>= 1) {
        if (c < o) s[c] += s[c + o];
        __syncthreads();
    }
    if (c == 0) {
        atomicAdd(loss, s[0] * (BETA / (float)((size_t)NQ * D)));
        idxf[n] = (float)k;
    }
}

// ---------------------------------------------------------------------------
extern "C" void kernel_launch(void* const* d_in, const int* in_sizes, int n_in,
                              void* d_out, int out_size, void* d_ws, size_t ws_size,
                              hipStream_t stream) {
    const float* z = (const float*)d_in[0];   // (16,128,32,32) fp32
    const float* w = (const float*)d_in[1];   // (8192,128) fp32, unit rows

    char* ws = (char*)d_ws;
    unsigned short* zbf = (unsigned short*)ws;                                 // 4 MB
    unsigned short* wbf = (unsigned short*)(ws + (size_t)NQ * D * 2);          // 2 MB
    float* csq   = (float*)(ws + (size_t)NQ * D * 2 + (size_t)K * D * 2);      // 32 KB
    float* bestv = (float*)((char*)csq + (size_t)K * sizeof(float));           // 128 KB
    int*   besti = (int*)((char*)bestv + (size_t)NSEG * NQ * sizeof(float));   // 128 KB

    float* out  = (float*)d_out;
    float* zq   = out;                        // 2097152 floats (B,C,H,W)
    float* loss = out + (size_t)NQ * D;       // 1 float
    float* idxf = loss + 1;                   // 16384 floats

    conv_weight<<<K,                128, 0, stream>>>(w, wbf, csq);
    norm_z     <<<NQ,               128, 0, stream>>>(z, zbf, loss);
    argmin_wmma<<<(NQ / 128) * NSEG, 256, 0, stream>>>(zbf, wbf, csq, bestv, besti);
    output_kernel<<<NQ,             128, 0, stream>>>(z, w, bestv, besti, zq, loss, idxf);
}
